// Net_58952721105460
// MI455X (gfx1250) — compile-verified, matrix operand order checked by reading, not run on hardware
//
#include <hip/hip_runtime.h>
#include <math.h>

// ---------------------------------------------------------------------------
// Resonate-and-fire SNN, fused: one kernel keeps (I1,V1) state for hidden cols
// 0..511 only (16 waves x 2 N-tiles) in WMMA accumulator registers across all 25 timesteps.
// bf16 WMMA (v_wmma_f32_16x16x32_bf16) for both GEMMs, f32 accumulation.
// K-loops kept ROLLED so LICM cannot hoist the (t-invariant) weight loads out
// of the timestep loop (that was the cause of the scratch spills); an asm
// memory clobber in the t-loop body blocks invariant-load hoisting entirely.
// M=32 per block halves W1 L2 traffic and gives 2x A/B fragment reuse.
// ---------------------------------------------------------------------------

typedef __attribute__((ext_vector_type(8)))  __bf16 v8bf;
typedef __attribute__((ext_vector_type(16))) __bf16 v16bf;
typedef __attribute__((ext_vector_type(8)))  float  v8f;

#define T_STEPS 25
#define BATCH_N 4096
#define N_IN    784
#define N_IN_P  800     // K padded to multiple of 32
#define N_HID   1000
#define N_HID_P 1024    // N padded to 64 tiles of 16
#define N_OUT   10
#define N_OUT_P 16
#define M_TILE  32      // batch rows per block (2 WMMA M-tiles)

// ---- weight pre-pack kernels (f32 -> zero-padded bf16, row-major) ----------

__global__ void pack_w1(const float* __restrict__ W1, __bf16* __restrict__ w1b) {
  int i = blockIdx.x * 256 + threadIdx.x;          // over 1024*800
  if (i >= N_HID_P * N_IN_P) return;
  int n = i / N_IN_P, k = i % N_IN_P;
  float v = (n < N_HID && k < N_IN) ? W1[n * N_IN + k] : 0.0f;
  w1b[i] = (__bf16)v;
}

__global__ void pack_w2(const float* __restrict__ W2, __bf16* __restrict__ w2b) {
  int i = blockIdx.x * 256 + threadIdx.x;          // over 16*1024
  if (i >= N_OUT_P * N_HID_P) return;
  int o = i / N_HID_P, h = i % N_HID_P;
  float v = (o < N_OUT && h < N_HID) ? W2[o * N_HID + h] : 0.0f;
  w2b[i] = (__bf16)v;
}

// ---- fragment helpers ------------------------------------------------------

__device__ inline v16bf load_frag(const __bf16* p) {
  // lane-local 16 bf16 = two 16B chunks (K x and K x+16), per ISA A/B layout
  v8bf lo = *(const v8bf*)(p);
  v8bf hi = *(const v8bf*)(p + 16);
  return __builtin_shufflevector(lo, hi, 0,1,2,3,4,5,6,7,8,9,10,11,12,13,14,15);
}

__device__ inline v8f vzero8() {
  v8f z;
#pragma unroll
  for (int i = 0; i < 8; ++i) z[i] = 0.0f;
  return z;
}

#define WMMA_BF16(a, b, c) \
  __builtin_amdgcn_wmma_f32_16x16x32_bf16(false, (a), false, (b), (short)0, (c), false, false)

// RAF phase-rotation + threshold + soft reset; writes spikes (stride-N_HID_P
// column of LDS) as bf16.
__device__ inline void raf_tile(v8f& I, v8f& V, const v8f& c, __bf16* p,
                                float Ac, float Bc) {
#pragma unroll
  for (int e = 0; e < 8; ++e) {
    float Io = I[e], Vo = V[e];
    float In = Ac * Io - Bc * Vo + c[e];
    float Vn = Bc * Io + Ac * Vo;
    float s  = (Vn > 1.0f) ? 1.0f : 0.0f;
    Vn -= s;
    I[e] = In; V[e] = Vn;
    p[(size_t)e * N_HID_P] = (__bf16)s;
  }
}

// ---- fused RAF network -----------------------------------------------------
// grid: 128 blocks (32 batch rows each), block: 512 threads = 16 waves.
// Wave w owns C-tiles (2 M-tiles) x (N-tiles {2w, 2w+1}); state in VGPRs.

__launch_bounds__(512, 1)
__global__ void raf_fused(const float* __restrict__ x,
                          const __bf16* __restrict__ w1b,
                          const __bf16* __restrict__ w2b,
                          float* __restrict__ out,
                          float Ac, float Bc) {
  __shared__ __bf16 lds_x[M_TILE][N_IN_P];     // 51.2 KB  staged x[t] tile
  __shared__ __bf16 lds_spk[M_TILE][N_HID_P];  // 65.5 KB  layer-1 spikes

  const int tid  = threadIdx.x;
  const int wave = tid >> 5;
  const int lane = tid & 31;
  const int half = lane >> 4;   // which K-half this lane holds in A/B frags
  const int lrow = lane & 15;   // row (A) / col (B,C) index within tile
  const int m0   = blockIdx.x * M_TILE;

  // fragment base pointers
  const __bf16* xap = &lds_x[lrow][half * 8];                            // A, layer 1 (mtile 0; +16 rows for mtile 1)
  const __bf16* w1p = w1b + (size_t)((wave * 2) * 16 + lrow) * N_IN_P + half * 8;  // B, layer 1
  const __bf16* sap = &lds_spk[(wave & 1) * 16 + lrow][half * 8];        // A, layer 2 (waves 0/1)
  const __bf16* w2p = w2b + (size_t)lrow * N_HID_P + half * 8;           // B, layer 2

  // zero the K-padding columns of lds_x once (never rewritten)
  if (tid < M_TILE * (N_IN_P - N_IN)) {
    int r = tid / (N_IN_P - N_IN), c = tid % (N_IN_P - N_IN);
    lds_x[r][N_IN + c] = (__bf16)0.0f;
  }

  // persistent RAF state in registers: [mtile][ntile]
  v8f I00 = vzero8(), V00 = vzero8(), I01 = vzero8(), V01 = vzero8();
  v8f I10 = vzero8(), V10 = vzero8(), I11 = vzero8(), V11 = vzero8();
  v8f I2 = vzero8(), V2 = vzero8();      // waves 0/1 only

  for (int t = 0; t < T_STEPS; ++t) {
    // opaque memory clobber: blocks LICM from hoisting the (otherwise
    // t-invariant) weight loads out of the timestep loop -> no mass spills
    asm volatile("" ::: "memory");
    __syncthreads();  // lds_spk free (waves 0/1 done), lds_x readers done

    // ---- stage x[t] tile: f32 global -> bf16 LDS -----------------------
    const float* xt = x + ((size_t)t * BATCH_N + m0) * N_IN;
    for (int i = tid; i < M_TILE * (N_IN / 4); i += 512) {
      int r = i / (N_IN / 4), c4 = i % (N_IN / 4);
      const float4 v = *(const float4*)(xt + r * N_IN + c4 * 4);
      __bf16* p = &lds_x[r][c4 * 4];
      p[0] = (__bf16)v.x; p[1] = (__bf16)v.y;
      p[2] = (__bf16)v.z; p[3] = (__bf16)v.w;
    }
    // prefetch next timestep's tile (2 x 128B per thread = 128KB >= 100KB)
    if (t + 1 < T_STEPS) {
      const char* xn = (const char*)(x + ((size_t)(t + 1) * BATCH_N + m0) * N_IN);
      __builtin_prefetch(xn + (size_t)tid * 128, 0, 3);
      __builtin_prefetch(xn + 65536 + (size_t)tid * 128, 0, 3);
    }
    __syncthreads();

    // ---- layer 1: cur1 = x_tile @ W1^T  (32 x 512, cols 0..511, K=800) -------------
    v8f c00 = vzero8(), c01 = vzero8(), c10 = vzero8(), c11 = vzero8();

#pragma unroll 1   // keep rolled: varying addresses defeat LICM hoisting
    for (int k0 = 0; k0 < N_IN_P; k0 += 32) {
      v16bf a0 = load_frag(xap + k0);                    // M rows 0..15
      v16bf a1 = load_frag(xap + 16 * N_IN_P + k0);      // M rows 16..31
      v16bf b0 = load_frag(w1p + k0);                    // N tile 2w
      v16bf b1 = load_frag(w1p + 16 * N_IN_P + k0);      // N tile 2w+1
      c00 = WMMA_BF16(a0, b0, c00);
      c01 = WMMA_BF16(a0, b1, c01);
      c10 = WMMA_BF16(a1, b0, c10);
      c11 = WMMA_BF16(a1, b1, c11);
    }

    // ---- RAF update layer 1, emit spikes to LDS ------------------------
    {
      int n0 = (wave * 2) * 16 + lrow;
      int n1 = (wave * 2 + 1) * 16 + lrow;
      raf_tile(I00, V00, c00, &lds_spk[half * 8][n0],      Ac, Bc);
      raf_tile(I01, V01, c01, &lds_spk[half * 8][n1],      Ac, Bc);
      raf_tile(I10, V10, c10, &lds_spk[16 + half * 8][n0], Ac, Bc);
      raf_tile(I11, V11, c11, &lds_spk[16 + half * 8][n1], Ac, Bc);
    }
    __syncthreads();

    // ---- layer 2 (waves 0,1): cur2 = spk1 @ W2^T  (16x16 each, K=1024) --
    if (wave < 2) {
      v8f acc2 = vzero8();
#pragma unroll 1
      for (int k0 = 0; k0 < N_HID_P; k0 += 32) {
        v16bf a = load_frag(sap + k0);
        v16bf b = load_frag(w2p + k0);
        acc2 = WMMA_BF16(a, b, acc2);
      }
#pragma unroll
      for (int e = 0; e < 8; ++e) {
        float Io = I2[e], Vo = V2[e];
        float In = Ac * Io - Bc * Vo + acc2[e];
        float Vn = Bc * Io + Ac * Vo;
        float s  = (Vn > 1.0f) ? 1.0f : 0.0f;
        Vn -= s;
        I2[e] = In; V2[e] = Vn;
        int m = wave * 16 + half * 8 + e;   // batch row within block tile
        int n = lrow;                       // output neuron
        if (n < N_OUT) {
          size_t idx = (size_t)t * BATCH_N * N_OUT + (size_t)(m0 + m) * N_OUT + n;
          out[idx] = s;                                          // spk2_rec
          out[idx + (size_t)T_STEPS * BATCH_N * N_OUT] = Vn;     // V2_rec
        }
      }
    }
  }
}

// ---------------------------------------------------------------------------

extern "C" void kernel_launch(void* const* d_in, const int* in_sizes, int n_in,
                              void* d_out, int out_size, void* d_ws, size_t ws_size,
                              hipStream_t stream) {
  const float* x  = (const float*)d_in[0];   // [25, 4096, 784]
  const float* W1 = (const float*)d_in[1];   // [1000, 784]
  const float* W2 = (const float*)d_in[2];   // [10, 1000]
  float* out = (float*)d_out;                // spk2_rec ++ V2_rec

  __bf16* w1b = (__bf16*)d_ws;                                   // 1024*800 bf16
  __bf16* w2b = (__bf16*)((char*)d_ws + (size_t)N_HID_P * N_IN_P * 2);

  pack_w1<<<(N_HID_P * N_IN_P + 255) / 256, 256, 0, stream>>>(W1, w1b);
  pack_w2<<<(N_OUT_P * N_HID_P + 255) / 256, 256, 0, stream>>>(W2, w2b);

  const float OMEGA = 2.0f * 3.14159265358979323846f * 30.0f * 1e-3f;
  const float Ac = 0.99f * cosf(OMEGA);
  const float Bc = 0.99f * sinf(OMEGA);

  raf_fused<<<BATCH_N / M_TILE, 512, 0, stream>>>(x, w1b, w2b, out, Ac, Bc);
}
